// EventMamba_18184891531856
// MI455X (gfx1250) — matrix operand order repaired and checked by reading, not compiled, not run on hardware
//
#include <hip/hip_runtime.h>
#include <math.h>

typedef __attribute__((ext_vector_type(16))) _Float16 v16h;
typedef __attribute__((ext_vector_type(8)))  float    v8f;

__device__ __forceinline__ v8f wmma_f16(v16h a, v16h b, v8f c) {
  return __builtin_amdgcn_wmma_f32_16x16x32_f16(false, a, false, b, (short)0, c, false, false);
}
__device__ __forceinline__ float sigmoidf_(float x){ return 1.f/(1.f+__expf(-x)); }
__device__ __forceinline__ float siluf_(float x){ return x*sigmoidf_(x); }
__device__ __forceinline__ float softplusf_(float x){ return (x>20.f)? x : log1pf(__expf(x)); }

// Branchless fragment loaders: indices are clamped so every load is legal and
// unconditional; invalid values are zeroed with selects (v_cndmask), not branches.
__device__ __forceinline__ v16h frag_a_full(const float* __restrict__ row, bool ok, int kb, int half) {
  v16h a;
#pragma unroll
  for (int e = 0; e < 16; ++e) {
    int ka = kb + e + half * 16;
    float v = row[ka];
    a[e] = (_Float16)(ok ? v : 0.f);
  }
  return a;
}
__device__ __forceinline__ v16h frag_a_tail(const float* __restrict__ row, bool ok, int kb, int half, int K) {
  v16h a;
#pragma unroll
  for (int e = 0; e < 16; ++e) {
    int ka = kb + e + half * 16;
    int kc = (ka < K) ? ka : (K - 1);
    float v = row[kc];
    a[e] = (_Float16)((ok && ka < K) ? v : 0.f);
  }
  return a;
}
__device__ __forceinline__ v16h frag_b_full(const float* __restrict__ row, bool ok, int kb, int half) {
  v16h b;
#pragma unroll
  for (int e = 0; e < 16; ++e) {
    int kk = kb + e + half * 16;
    float v = row[kk];
    b[e] = (_Float16)(ok ? v : 0.f);
  }
  return b;
}
__device__ __forceinline__ v16h frag_b_tail(const float* __restrict__ row, bool ok, int kb, int half, int K) {
  v16h b;
#pragma unroll
  for (int e = 0; e < 16; ++e) {
    int kk = kb + e + half * 16;
    int kc = (kk < K) ? kk : (K - 1);
    float v = row[kc];
    b[e] = (_Float16)((ok && kk < K) ? v : 0.f);
  }
  return b;
}

// ---------------------------------------------------------------- GEMM (WMMA)
// Y(M,N) = act( bn( X(M,K) @ W(N,K)^T + bias ) + resid )
__global__ void gemm_wmma_kernel(const float* __restrict__ X, int ldX,
                                 const float* __restrict__ Wt,
                                 const float* __restrict__ bias,
                                 const float* __restrict__ bng, const float* __restrict__ bnb,
                                 const float* __restrict__ bnrm, const float* __restrict__ bnrv,
                                 const float* __restrict__ resid,
                                 float* __restrict__ Y, int ldY,
                                 int M, int N, int K, int act)
{
  int Nt = (N + 15) >> 4;
  int Mt = (M + 15) >> 4;
  int wave = blockIdx.x * (blockDim.x >> 5) + (threadIdx.x >> 5);
  if (wave >= Mt * Nt) return;
  int mt = wave / Nt, nt = wave % Nt;
  int lane = threadIdx.x & 31;
  int half = lane >> 4, r = lane & 15;
  int arow = mt * 16 + r;
  int bcol = nt * 16 + r;
  bool aok = arow < M;
  bool bok = bcol < N;
  const float* Xr = X + (size_t)(aok ? arow : (M - 1)) * ldX;
  const float* Wr = Wt + (size_t)(bok ? bcol : (N - 1)) * K;
  v8f acc = {};
  int Kfull = K & ~31;
  for (int kb = 0; kb < Kfull; kb += 32) {
    v16h a = frag_a_full(Xr, aok, kb, half);
    v16h bf = frag_b_full(Wr, bok, kb, half);
    acc = wmma_f16(a, bf, acc);
  }
  if (Kfull < K) {
    v16h a = frag_a_tail(Xr, aok, Kfull, half, K);
    v16h bf = frag_b_tail(Wr, bok, Kfull, half, K);
    acc = wmma_f16(a, bf, acc);
  }
#pragma unroll
  for (int j = 0; j < 8; ++j) {
    int row = mt * 16 + j + half * 8;
    int col = nt * 16 + r;
    if (row < M && col < N) {
      float y = acc[j];
      if (bias) y += bias[col];
      if (bng)  y = (y - bnrm[col]) * rsqrtf(bnrv[col] + 1e-5f) * bng[col] + bnb[col];
      if (resid) y += resid[(size_t)row * ldY + col];
      if (act == 1) y = fmaxf(y, 0.f);
      else if (act == 2) y = softplusf_(y);
      Y[(size_t)row * ldY + col] = y;
    }
  }
}

// ---------------------------------------------------------------- fused group MLP + att pool
struct FusedArgs {
  const float *pts, *mean, *invstd, *alpha, *beta;
  const int* knn;
  const float *pw,*pb,*pg,*pbb,*prm,*prv;     // optional pre LIN1
  const float *w1,*b1,*g1,*bb1,*rm1,*rv1;     // LIN2 first
  const float *w2,*b2,*g2,*bb2,*rm2,*rv2;     // LIN2 second
  const float *aw,*ab;                        // attention
  float* fout;
  int Nsrc, G, Cin, C2, C, hasPre;
};

// M=24 rows fixed; src in LDS row-major width K; W (N,K) in global (N mult of 16).
__device__ void mlp24_wmma(const float* src, int K, const float* __restrict__ Wt,
                           const float* bias,
                           const float* bng, const float* bnb,
                           const float* bnrm, const float* bnrv,
                           const float* resid, float* dst, int N, int relu)
{
  int lane = threadIdx.x & 31, wv = threadIdx.x >> 5;
  int half = lane >> 4, r = lane & 15;
  int Nt = N >> 4;
  for (int tile = wv; tile < 2 * Nt; tile += 8) {
    int mt = tile & 1, nt = tile >> 1;
    int arow = mt * 16 + r;
    int bcol = nt * 16 + r;
    bool aok = arow < 24;
    const float* Sr = src + (aok ? arow : 23) * K;
    const float* Wr = Wt + (size_t)bcol * K;
    v8f acc = {};
    int Kfull = K & ~31;
    for (int kb = 0; kb < Kfull; kb += 32) {
      v16h a = frag_a_full(Sr, aok, kb, half);
      v16h bf = frag_b_full(Wr, true, kb, half);
      acc = wmma_f16(a, bf, acc);
    }
    if (Kfull < K) {
      v16h a = frag_a_tail(Sr, aok, Kfull, half, K);
      v16h bf = frag_b_tail(Wr, true, Kfull, half, K);
      acc = wmma_f16(a, bf, acc);
    }
#pragma unroll
    for (int j = 0; j < 8; ++j) {
      int row = mt * 16 + j + half * 8;
      int col = nt * 16 + r;
      if (row < 24) {
        float y = acc[j] + (bias ? bias[col] : 0.f);
        if (bng) y = (y - bnrm[col]) * rsqrtf(bnrv[col] + 1e-5f) * bng[col] + bnb[col];
        if (resid) y += resid[row * N + col];
        if (relu) y = fmaxf(y, 0.f);
        dst[row * N + col] = y;
      }
    }
  }
  __syncthreads();
}

__global__ void fused_group_kernel(FusedArgs A)
{
  extern __shared__ float smem[];
  float* s_x = smem;                  // 24*C
  float* s_y = s_x + 24 * A.C;        // 24*C
  float* s_t = s_y + 24 * A.C;        // 24*(C/2)
  __shared__ float s_sc[32];
  int g = blockIdx.x, b = blockIdx.y;
  int tid = threadIdx.x;
  int Cin = A.Cin, C2 = A.C2, C = A.C;
  const int* kn = A.knn + ((size_t)b * A.G + g) * 24;
  const float* mn = A.mean + ((size_t)b * A.G + g) * Cin;
  float istd = A.invstd[b];
  for (int e = tid; e < 24 * C2; e += blockDim.x) {
    int j = e / C2, c = e % C2;
    float v;
    if (c < Cin) {
      float p = A.pts[((size_t)b * A.Nsrc + kn[j]) * Cin + c];
      v = A.alpha[c] * ((p - mn[c]) * istd) + A.beta[c];
    } else v = mn[c - Cin];
    s_x[j * C2 + c] = v;
  }
  __syncthreads();
  const float* fin0;
  if (A.hasPre) {
    mlp24_wmma(s_x, C2, A.pw, A.pb, A.pg, A.pbb, A.prm, A.prv, nullptr, s_y, C, 1);
    fin0 = s_y;
  } else fin0 = s_x;
  float* out2 = (fin0 == s_y) ? s_x : s_y;
  mlp24_wmma(fin0, C, A.w1, A.b1, A.g1, A.bb1, A.rm1, A.rv1, nullptr, s_t, C / 2, 1);
  mlp24_wmma(s_t, C / 2, A.w2, A.b2, A.g2, A.bb2, A.rm2, A.rv2, fin0, out2, C, 1);
  // attention pool over 24 rows
  if (tid < 24) {
    float s = A.ab[0];
    for (int c = 0; c < C; ++c) s += out2[tid * C + c] * A.aw[c];
    s_sc[tid] = s;
  }
  __syncthreads();
  if (tid == 0) {
    float m = -1e30f;
    for (int j = 0; j < 24; ++j) m = fmaxf(m, s_sc[j]);
    float sum = 0.f;
    for (int j = 0; j < 24; ++j) { float eV = __expf(s_sc[j] - m); s_sc[j] = eV; sum += eV; }
    float inv = 1.f / sum;
    for (int j = 0; j < 24; ++j) s_sc[j] *= inv;
  }
  __syncthreads();
  for (int c = tid; c < C; c += blockDim.x) {
    float acc = 0.f;
    for (int j = 0; j < 24; ++j) acc += s_sc[j] * out2[j * C + c];
    A.fout[((size_t)b * A.G + g) * C + c] = acc;
  }
}

// ---------------------------------------------------------------- FPS
__global__ void fps_kernel(const float* __restrict__ xyz, int* __restrict__ out, int N, int npoint)
{
  int b = blockIdx.x, tid = threadIdx.x;
  __shared__ float dist[1024];
  __shared__ float redv[256];
  __shared__ int   redi[256];
  __shared__ float cen[3];
  __shared__ int far_s;
  const float* base = xyz + (size_t)b * N * 3;
  for (int i = tid; i < N; i += blockDim.x) dist[i] = 1e10f;
  if (tid == 0) far_s = 0;
  __syncthreads();
  for (int it = 0; it < npoint; ++it) {
    if (tid == 0) {
      out[(size_t)b * npoint + it] = far_s;
      cen[0] = base[far_s * 3 + 0];
      cen[1] = base[far_s * 3 + 1];
      cen[2] = base[far_s * 3 + 2];
    }
    __syncthreads();
    float best = -1.f; int bidx = 0;
    for (int i = tid; i < N; i += blockDim.x) {
      float dx = base[i*3+0]-cen[0], dy = base[i*3+1]-cen[1], dz = base[i*3+2]-cen[2];
      float d = dx*dx + dy*dy + dz*dz;
      float nd = fminf(dist[i], d);
      dist[i] = nd;
      if (nd > best) { best = nd; bidx = i; }
    }
    redv[tid] = best; redi[tid] = bidx;
    __syncthreads();
    for (int s = blockDim.x >> 1; s > 0; s >>= 1) {
      if (tid < s) {
        float ov = redv[tid+s]; int oi = redi[tid+s];
        if (ov > redv[tid] || (ov == redv[tid] && oi < redi[tid])) { redv[tid]=ov; redi[tid]=oi; }
      }
      __syncthreads();
    }
    if (tid == 0) far_s = redi[0];
    __syncthreads();
  }
}

// ---------------------------------------------------------------- KNN (k=24)
__global__ void knn_kernel(const float* __restrict__ xyz, const float* __restrict__ q,
                           int* __restrict__ out, int B, int N, int G)
{
  int t = blockIdx.x * blockDim.x + threadIdx.x;
  if (t >= B * G) return;
  int b = t / G;
  const float* base = xyz + (size_t)b * N * 3;
  float qx = q[(size_t)t*3], qy = q[(size_t)t*3+1], qz = q[(size_t)t*3+2];
  float kd[24]; int ki[24];
  for (int j = 0; j < 24; ++j) { kd[j] = 1e30f; ki[j] = 0; }
  for (int i = 0; i < N; ++i) {
    float dx = base[i*3]-qx, dy = base[i*3+1]-qy, dz = base[i*3+2]-qz;
    float d = dx*dx + dy*dy + dz*dz;
    if (d < kd[23]) {
      int p = 23;
      while (p > 0 && kd[p-1] > d) { kd[p] = kd[p-1]; ki[p] = ki[p-1]; --p; }
      kd[p] = d; ki[p] = i;
    }
  }
  for (int j = 0; j < 24; ++j) out[(size_t)t * 24 + j] = ki[j];
}

// ---------------------------------------------------------------- misc small kernels
__global__ void zero_kernel(float* p, int n){ int i = blockIdx.x*blockDim.x+threadIdx.x; if(i<n) p[i]=0.f; }

__global__ void stats_kernel(const float* __restrict__ pts, const int* __restrict__ knn,
                             const float* __restrict__ mean, float* __restrict__ stats,
                             int Nsrc, int G, int Cin)
{
  int b = blockIdx.y;
  int elems = G * 24 * Cin;
  float s = 0.f, s2 = 0.f;
  for (int e = blockIdx.x * blockDim.x + threadIdx.x; e < elems; e += gridDim.x * blockDim.x) {
    int g = e / (24 * Cin);
    int rem = e % (24 * Cin);
    int j = rem / Cin, c = rem % Cin;
    int idx = knn[((size_t)b * G + g) * 24 + j];
    float d = pts[((size_t)b * Nsrc + idx) * Cin + c] - mean[((size_t)b * G + g) * Cin + c];
    s += d; s2 += d * d;
  }
  __shared__ float r1[256], r2[256];
  r1[threadIdx.x] = s; r2[threadIdx.x] = s2;
  __syncthreads();
  for (int st = blockDim.x >> 1; st > 0; st >>= 1) {
    if (threadIdx.x < st) { r1[threadIdx.x] += r1[threadIdx.x+st]; r2[threadIdx.x] += r2[threadIdx.x+st]; }
    __syncthreads();
  }
  if (threadIdx.x == 0) { atomicAdd(&stats[b*2], r1[0]); atomicAdd(&stats[b*2+1], r2[0]); }
}

__global__ void invstd_kernel(const float* stats, float* invstd, int B, int nelem)
{
  int b = blockIdx.x*blockDim.x + threadIdx.x;
  if (b < B) {
    float s = stats[b*2], s2 = stats[b*2+1];
    float n = (float)nelem;
    float var = (s2 - s*s/n) / (n - 1.f);
    float sd = sqrtf(fmaxf(var, 0.f));
    invstd[b] = 1.f / (sd + 1e-5f);
  }
}

__global__ void layernorm_kernel(const float* __restrict__ x, const float* __restrict__ g,
                                 const float* __restrict__ b, float* __restrict__ y, int d)
{
  int row = blockIdx.x, tid = threadIdx.x;
  const float* xr = x + (size_t)row * d;
  __shared__ float r1[256], r2[256];
  float s=0.f, s2=0.f;
  for (int c = tid; c < d; c += blockDim.x) { float v = xr[c]; s += v; s2 += v*v; }
  r1[tid]=s; r2[tid]=s2; __syncthreads();
  for (int st = blockDim.x>>1; st>0; st>>=1){ if(tid<st){r1[tid]+=r1[tid+st]; r2[tid]+=r2[tid+st];} __syncthreads(); }
  float mean = r1[0]/d;
  float var = r2[0]/d - mean*mean;
  float inv = rsqrtf(fmaxf(var,0.f)+1e-5f);
  for (int c = tid; c < d; c += blockDim.x)
    y[(size_t)row*d + c] = (xr[c]-mean)*inv*g[c] + b[c];
}

__global__ void conv_silu_kernel(const float* __restrict__ src, int lds, int off,
                                 const float* __restrict__ w, const float* __restrict__ bias,
                                 float* __restrict__ out, int B, int G, int D)
{
  size_t total = (size_t)B*G*D;
  for (size_t e = (size_t)blockIdx.x*blockDim.x+threadIdx.x; e < total; e += (size_t)gridDim.x*blockDim.x) {
    int d = (int)(e % D); size_t bt = e / D; int t = (int)(bt % G); int b = (int)(bt / G);
    float acc = bias[d];
#pragma unroll
    for (int j = 0; j < 4; ++j) {
      int tt = t - 3 + j;
      if (tt >= 0) acc += src[((size_t)b*G + tt)*lds + off + d] * w[d*4 + j];
    }
    out[((size_t)b*G + t)*D + d] = siluf_(acc);
  }
}

__global__ void flip_kernel(const float* __restrict__ src, int lds, int off,
                            float* __restrict__ dst, int C, int B, int G)
{
  size_t total = (size_t)B*G*C;
  for (size_t e = (size_t)blockIdx.x*blockDim.x+threadIdx.x; e < total; e += (size_t)gridDim.x*blockDim.x) {
    int c = (int)(e % C); size_t bt = e / C; int t = (int)(bt % G); int b = (int)(bt / G);
    dst[((size_t)b*G+t)*C + c] = src[((size_t)b*G + (G-1-t))*lds + off + c];
  }
}

__global__ void add_flip_kernel(const float* __restrict__ a, const float* __restrict__ bfl,
                                float* __restrict__ dst, int C, int B, int G)
{
  size_t total = (size_t)B*G*C;
  for (size_t e = (size_t)blockIdx.x*blockDim.x+threadIdx.x; e < total; e += (size_t)gridDim.x*blockDim.x) {
    int c = (int)(e % C); size_t bt = e / C; int t = (int)(bt % G); int b = (int)(bt / G);
    dst[e] = a[e] + bfl[((size_t)b*G + (G-1-t))*C + c];
  }
}

__global__ void ssm_scan_kernel(const float* __restrict__ dt, const float* __restrict__ dbl,
                                int ldb, int offB, int offC,
                                const float* __restrict__ xc,
                                const float* __restrict__ z, int ldz, int offz,
                                const float* __restrict__ A_log, const float* __restrict__ Dv,
                                float* __restrict__ y, int G, int Dd)
{
  int d = blockIdx.x*blockDim.x + threadIdx.x;
  int b = blockIdx.y;
  if (d >= Dd) return;
  float A[16], h[16];
#pragma unroll
  for (int n = 0; n < 16; ++n) { A[n] = -__expf(A_log[d*16+n]); h[n] = 0.f; }
  float Dval = Dv[d];
  for (int t = 0; t < G; ++t) {
    size_t row = (size_t)b*G + t;
    float dtv = dt[row*Dd + d];
    float xv  = xc[row*Dd + d];
    float acc = 0.f;
#pragma unroll
    for (int n = 0; n < 16; ++n) {
      float Bn = dbl[row*ldb + offB + n];
      float Cn = dbl[row*ldb + offC + n];
      h[n] = __expf(dtv*A[n])*h[n] + dtv*Bn*xv;
      acc += h[n]*Cn;
    }
    float yv = acc + xv*Dval;
    float zv = z[row*ldz + offz + d];
    y[row*Dd + d] = yv * siluf_(zv);
  }
}

__global__ void transpose_kernel(const float* __restrict__ x, float* __restrict__ xyz, int Np, int B)
{
  size_t total = (size_t)B*Np*3;
  for (size_t e = (size_t)blockIdx.x*blockDim.x+threadIdx.x; e < total; e += (size_t)gridDim.x*blockDim.x) {
    int c = (int)(e % 3); size_t bn = e/3; int n = (int)(bn % Np); int b = (int)(bn / Np);
    xyz[((size_t)b*Np+n)*3+c] = x[(size_t)b*3*Np + (size_t)c*Np + n];
  }
}

__global__ void gather_kernel(const float* __restrict__ src, const int* __restrict__ idx,
                              float* __restrict__ dst, int Nsrc, int G, int C, int B)
{
  size_t total = (size_t)B*G*C;
  for (size_t e = (size_t)blockIdx.x*blockDim.x+threadIdx.x; e < total; e += (size_t)gridDim.x*blockDim.x) {
    int c = (int)(e % C); size_t bg = e / C; int g = (int)(bg % G); int b = (int)(bg / G);
    dst[((size_t)b*G+g)*C + c] = src[((size_t)b*Nsrc + idx[(size_t)b*G+g])*C + c];
  }
}

__global__ void att_pool_kernel(const float* __restrict__ x, const float* __restrict__ aw,
                                const float* __restrict__ ab, float* __restrict__ out, int L, int C)
{
  int s = blockIdx.x, tid = threadIdx.x;
  extern __shared__ float sc[];
  const float* xb = x + (size_t)s * L * C;
  for (int l = tid; l < L; l += blockDim.x) {
    float v = ab[0];
    for (int c = 0; c < C; ++c) v += xb[(size_t)l*C+c]*aw[c];
    sc[l] = v;
  }
  __syncthreads();
  if (tid == 0) {
    float m = -1e30f; for (int l=0;l<L;++l) m = fmaxf(m, sc[l]);
    float sum=0.f; for (int l=0;l<L;++l){ float e=__expf(sc[l]-m); sc[l]=e; sum+=e; }
    float inv=1.f/sum; for (int l=0;l<L;++l) sc[l]*=inv;
  }
  __syncthreads();
  for (int c = tid; c < C; c += blockDim.x) {
    float acc=0.f; for (int l=0;l<L;++l) acc += sc[l]*xb[(size_t)l*C+c];
    out[(size_t)s*C + c] = acc;
  }
}

// ---------------------------------------------------------------- host side
struct BNp { const float *g,*b,*rm,*rv; };
struct L1p { const float *w,*b; BNp bn; };
struct L2p { const float *w1,*b1; BNp bn1; const float *w2,*b2; BNp bn2; };
struct ATTp { const float *w,*b; };
struct DIRp { const float *conv_w,*conv_b,*xproj_w,*dt_w,*dt_b,*A_log,*D; };
struct MBp  { const float *ln_g,*ln_b,*in_w,*out_w; DIRp fwd, bwd; };

extern "C" void kernel_launch(void* const* d_in, const int* in_sizes, int n_in,
                              void* d_out, int out_size, void* d_ws, size_t ws_size,
                              hipStream_t stream)
{
  if (n_in < 155) return;
  int ip = 0;
  auto nx  = [&]() { return (const float*)d_in[ip++]; };
  auto nbn = [&]() { BNp r; r.g=nx(); r.b=nx(); r.rm=nx(); r.rv=nx(); return r; };
  auto nl1 = [&]() { L1p r; r.w=nx(); r.b=nx(); r.bn=nbn(); return r; };
  auto nl2 = [&]() { L2p r; r.w1=nx(); r.b1=nx(); r.bn1=nbn(); r.w2=nx(); r.b2=nx(); r.bn2=nbn(); return r; };
  auto nat = [&]() { ATTp r; r.w=nx(); r.b=nx(); return r; };
  auto ndi = [&]() { DIRp r; r.conv_w=nx(); r.conv_b=nx(); r.xproj_w=nx(); r.dt_w=nx(); r.dt_b=nx(); r.A_log=nx(); r.D=nx(); return r; };
  auto nmb = [&]() { MBp r; r.ln_g=nx(); r.ln_b=nx(); r.in_w=nx(); r.out_w=nx(); r.fwd=ndi(); r.bwd=ndi(); return r; };

  const float* x    = nx();
  const float* g0_a = nx(); const float* g0_b = nx();
  const float* g1_a = nx(); const float* g1_b = nx();
  const float* g2_a = nx(); const float* g2_b = nx();
  L1p embed = nl1();
  L2p conv1 = nl2(), conv1_1 = nl2();
  L2p conv2 = nl2(), conv2_1 = nl2();
  L2p conv3 = nl2(), conv3_1 = nl2();
  ATTp att1 = nat(), att2 = nat(), att3 = nat(), att4 = nat();
  MBp m1 = nmb(), m2 = nmb(), m3 = nmb();
  const float* cls_w1 = nx(); const float* cls_b1 = nx();
  BNp cls_bn = nbn();
  const float* cls_w2 = nx(); const float* cls_b2 = nx();

  // workspace bump allocator
  float* Wbase = (float*)d_ws;
  size_t off = 0;
  auto alloc = [&](size_t n){ float* p = Wbase + off; off += n; return p; };
  const int B = 32;
  const size_t U = 1048576;   // B*G*C is 1M floats for every stage
  float* xyz0   = alloc((size_t)B*1024*3);
  float* nxyz1  = alloc((size_t)B*512*3);
  float* nxyz2  = alloc((size_t)B*256*3);
  float* nxyz3  = alloc((size_t)B*128*3);
  int*   fpsBuf = (int*)alloc((size_t)B*512);
  int*   knnBuf = (int*)alloc((size_t)B*512*24);
  float* meanBuf= alloc(524288);
  float* stats  = alloc(64);
  float* invstd = alloc(32);
  float* fPrev  = alloc(U);
  float* fPool  = alloc(U);
  float* fM     = alloc(U);
  float* tmpHalf= alloc(U/2);
  float* lnB    = alloc(U);
  float* xzB    = alloc(4*U);
  float* xfwB   = alloc(2*U);
  float* zwB    = alloc(2*U);
  float* xcB    = alloc(2*U);
  float* dblB   = alloc((size_t)16384*48);
  float* dtB    = alloc(2*U);
  float* ofB    = alloc(2*U);
  float* obB    = alloc(2*U);
  float* smB    = alloc(2*U);
  float* attout = alloc((size_t)B*256);
  float* clsh   = alloc((size_t)B*256);
  if (off * sizeof(float) > ws_size) return;

  auto gemm = [&](const float* X, int ldX, const float* Wt, const float* bias,
                  const float* bg, const float* bb, const float* brm, const float* brv,
                  const float* resid, float* Y, int ldY, int M, int N, int K, int act){
    int tiles = ((M+15)/16)*((N+15)/16);
    int blocks = (tiles + 7) / 8;
    gemm_wmma_kernel<<<blocks, 256, 0, stream>>>(X, ldX, Wt, bias, bg, bb, brm, brv,
                                                 resid, Y, ldY, M, N, K, act);
  };

  auto run_mamba = [&](const MBp& m, const float* fin, float* fout, int G, int d){
    int BG = B*G, di = 2*d, R = d/16; if (R < 1) R = 1; int R32 = R + 32;
    size_t tot = (size_t)BG*di; int blk = (int)((tot+255)/256);
    layernorm_kernel<<<BG, 256, 0, stream>>>(fin, m.ln_g, m.ln_b, lnB, d);
    gemm(lnB, d, m.in_w, nullptr, nullptr,nullptr,nullptr,nullptr, nullptr, xzB, 4*d, BG, 4*d, d, 0);
    // forward direction
    conv_silu_kernel<<<blk,256,0,stream>>>(xzB, 4*d, 0, m.fwd.conv_w, m.fwd.conv_b, xcB, B, G, di);
    gemm(xcB, di, m.fwd.xproj_w, nullptr, nullptr,nullptr,nullptr,nullptr, nullptr, dblB, R32, BG, R32, di, 0);
    gemm(dblB, R32, m.fwd.dt_w, m.fwd.dt_b, nullptr,nullptr,nullptr,nullptr, nullptr, dtB, di, BG, di, R, 2);
    ssm_scan_kernel<<<dim3((di+255)/256, B),256,0,stream>>>(dtB, dblB, R32, R, R+16,
        xcB, xzB, 4*d, di, m.fwd.A_log, m.fwd.D, ofB, G, di);
    // backward direction (flip sequence)
    flip_kernel<<<blk,256,0,stream>>>(xzB, 4*d, 0,  xfwB, di, B, G);
    flip_kernel<<<blk,256,0,stream>>>(xzB, 4*d, di, zwB,  di, B, G);
    conv_silu_kernel<<<blk,256,0,stream>>>(xfwB, di, 0, m.bwd.conv_w, m.bwd.conv_b, xcB, B, G, di);
    gemm(xcB, di, m.bwd.xproj_w, nullptr, nullptr,nullptr,nullptr,nullptr, nullptr, dblB, R32, BG, R32, di, 0);
    gemm(dblB, R32, m.bwd.dt_w, m.bwd.dt_b, nullptr,nullptr,nullptr,nullptr, nullptr, dtB, di, BG, di, R, 2);
    ssm_scan_kernel<<<dim3((di+255)/256, B),256,0,stream>>>(dtB, dblB, R32, R, R+16,
        xcB, zwB, di, 0, m.bwd.A_log, m.bwd.D, obB, G, di);
    add_flip_kernel<<<blk,256,0,stream>>>(ofB, obB, smB, di, B, G);
    gemm(smB, di, m.out_w, nullptr, nullptr,nullptr,nullptr,nullptr, fin, fout, d, BG, d, di, 0);
  };

  auto run_stage = [&](const float* srcXyz, int N, int G, const float* pts, int Cin,
                       const float* alpha, const float* beta,
                       const L1p* pre, const L2p& convp, const ATTp& att, const MBp& mb,
                       const L2p& outp, float* outXyz, float* featOut, int C)
  {
    int C2 = 2*Cin;
    fps_kernel<<<B, 256, 0, stream>>>(srcXyz, fpsBuf, N, G);
    { size_t t=(size_t)B*G*3; gather_kernel<<<(int)((t+255)/256),256,0,stream>>>(srcXyz, fpsBuf, outXyz, N, G, 3, B); }
    const float* meanPtr;
    if (pts == srcXyz && Cin == 3) meanPtr = outXyz;
    else {
      size_t t=(size_t)B*G*Cin;
      gather_kernel<<<(int)((t+255)/256),256,0,stream>>>(pts, fpsBuf, meanBuf, N, G, Cin, B);
      meanPtr = meanBuf;
    }
    knn_kernel<<<(B*G+255)/256, 256, 0, stream>>>(srcXyz, outXyz, knnBuf, B, N, G);
    zero_kernel<<<1,64,0,stream>>>(stats, 64);
    { int blocks = (G*24*Cin+255)/256; if (blocks > 256) blocks = 256;
      stats_kernel<<<dim3(blocks,B),256,0,stream>>>(pts, knnBuf, meanPtr, stats, N, G, Cin); }
    invstd_kernel<<<1,32,0,stream>>>(stats, invstd, B, G*24*Cin);
    FusedArgs fa{};
    fa.pts=pts; fa.mean=meanPtr; fa.invstd=invstd; fa.alpha=alpha; fa.beta=beta; fa.knn=knnBuf;
    if (pre){ fa.pw=pre->w; fa.pb=pre->b; fa.pg=pre->bn.g; fa.pbb=pre->bn.b; fa.prm=pre->bn.rm; fa.prv=pre->bn.rv; fa.hasPre=1; }
    fa.w1=convp.w1; fa.b1=convp.b1; fa.g1=convp.bn1.g; fa.bb1=convp.bn1.b; fa.rm1=convp.bn1.rm; fa.rv1=convp.bn1.rv;
    fa.w2=convp.w2; fa.b2=convp.b2; fa.g2=convp.bn2.g; fa.bb2=convp.bn2.b; fa.rm2=convp.bn2.rm; fa.rv2=convp.bn2.rv;
    fa.aw=att.w; fa.ab=att.b; fa.fout=fPool;
    fa.Nsrc=N; fa.G=G; fa.Cin=Cin; fa.C2=C2; fa.C=C;
    size_t smem = (size_t)(24*C*2 + 24*(C/2)) * sizeof(float);
    fused_group_kernel<<<dim3(G,B),256,smem,stream>>>(fa);
    run_mamba(mb, fPool, fM, G, C);
    gemm(fM, C, outp.w1, outp.b1, outp.bn1.g, outp.bn1.b, outp.bn1.rm, outp.bn1.rv,
         nullptr, tmpHalf, C/2, B*G, C/2, C, 1);
    gemm(tmpHalf, C/2, outp.w2, outp.b2, outp.bn2.g, outp.bn2.b, outp.bn2.rm, outp.bn2.rv,
         fM, featOut, C, B*G, C, C/2, 1);
  };

  // x (B,3,1024) -> xyz (B,1024,3)
  { size_t t=(size_t)B*1024*3; transpose_kernel<<<(int)((t+255)/256),256,0,stream>>>(x, xyz0, 1024, B); }

  run_stage(xyz0,  1024, 512, xyz0,  3,   g0_a, g0_b, &embed,  conv1, att1, m1, conv1_1, nxyz1, fPrev, 64);
  run_stage(nxyz1, 512,  256, fPrev, 64,  g1_a, g1_b, nullptr, conv2, att2, m2, conv2_1, nxyz2, fPrev, 128);
  run_stage(nxyz2, 256,  128, fPrev, 128, g2_a, g2_b, nullptr, conv3, att3, m3, conv3_1, nxyz3, fPrev, 256);

  // classifier head
  att_pool_kernel<<<B, 256, 128*sizeof(float), stream>>>(fPrev, att4.w, att4.b, attout, 128, 256);
  gemm(attout, 256, cls_w1, cls_b1, cls_bn.g, cls_bn.b, cls_bn.rm, cls_bn.rv, nullptr, clsh, 256, B, 256, 256, 1);
  gemm(clsh, 256, cls_w2, cls_b2, nullptr,nullptr,nullptr,nullptr, nullptr, (float*)d_out, 6, B, 6, 256, 0);
}
